// CausalSelfAttention_71631464563127
// MI455X (gfx1250) — matrix operand order repaired and checked by reading, not compiled, not run on hardware
//
#include <hip/hip_runtime.h>

// ---------------------------------------------------------------------------
// CDNA5 / gfx1250 causal self-attention, bf16 WMMA pipeline (round 2).
//   x[4096,1024] @ Wqkv[1024,3072] -> Q,K,V [B=2,H=16,T=2048,Dh=64] (bf16)
//   flash-attn (4 waves/block share K/V tiles) with v_wmma_f32_16x16x32_bf16
//   attn_out[4096,1024](bf16) @ Wout[1024,1024] -> d_out fp32
// ---------------------------------------------------------------------------

typedef __attribute__((ext_vector_type(16))) __bf16 bf16x16;
typedef __attribute__((ext_vector_type(8)))  float  floatx8;

union FragB16 { bf16x16 v; unsigned short u[16]; unsigned int w[8]; uint4 q[2]; };
union FragF8  { floatx8 v; float f[8]; };

static __device__ __forceinline__ unsigned short f2bf(float f) {
    unsigned u = __builtin_bit_cast(unsigned, f);
    unsigned r = (u + 0x7FFFu + ((u >> 16) & 1u)) >> 16;
    return (unsigned short)r;
}

// ------------------------------ fp32 -> bf16 -------------------------------
__global__ __launch_bounds__(256) void cvt_f32_bf16(const float* __restrict__ in,
                                                    unsigned short* __restrict__ out,
                                                    int n) {
    int i = blockIdx.x * 256 + threadIdx.x;
    if (i < n) out[i] = f2bf(in[i]);
}

// ------------------------------ WMMA GEMM ----------------------------------
// Block: 256 threads = 8 waves in a 2x4 grid. Block tile 64(M) x 128(N).
// Each wave owns a 32x32 macro-tile => 4 WMMAs per 32-K step (frag reuse).
// MODE 0: C fp32 row-major [M,N].
// MODE 1: scatter bf16 into Q/K/V [B,H,T,64].
template <int MODE>
__global__ __launch_bounds__(256) void gemm_bf16_wmma(
    const unsigned short* __restrict__ A,   // [M,K] bf16
    const unsigned short* __restrict__ B,   // [K,N] bf16
    float* __restrict__ C,
    unsigned short* __restrict__ Qb,
    unsigned short* __restrict__ Kb,
    unsigned short* __restrict__ Vb,
    int M, int N, int K)
{
    // pad strides keep rows 16B-aligned (uint4 stores) and rotate banks
    __shared__ unsigned short As[64][40];    // 64x32 tile, stride 40 (80B)
    __shared__ unsigned short Bs[32][136];   // 32x128 tile, stride 136 (272B)

    const int tid  = threadIdx.x;
    const int lane = tid & 31;
    const int w    = tid >> 5;          // 0..7
    const int wm   = w >> 2;            // 0..1  (32-row band)
    const int wn   = w & 3;             // 0..3  (32-col band)
    const int hh   = lane >> 4;
    const int l15  = lane & 15;

    const int m0 = blockIdx.x * 64;
    const int n0 = blockIdx.y * 128;

    FragF8 acc[2][2];
    #pragma unroll
    for (int mi = 0; mi < 2; ++mi)
        #pragma unroll
        for (int ni = 0; ni < 2; ++ni)
            #pragma unroll
            for (int r = 0; r < 8; ++r) acc[mi][ni].f[r] = 0.f;

    for (int k0 = 0; k0 < K; k0 += 32) {
        // ---- stage A: 64x32 bf16 = 256 uint4, 1 per thread ----
        {
            int row = tid >> 2, c = (tid & 3) * 8;
            *reinterpret_cast<uint4*>(&As[row][c]) =
                *reinterpret_cast<const uint4*>(&A[(size_t)(m0 + row) * K + (k0 + c)]);
        }
        // ---- stage B: 32x128 bf16 = 512 uint4, 2 per thread ----
        #pragma unroll
        for (int j = 0; j < 2; ++j) {
            int idx = tid + 256 * j;
            int row = idx >> 4, c = (idx & 15) * 8;
            *reinterpret_cast<uint4*>(&Bs[row][c]) =
                *reinterpret_cast<const uint4*>(&B[(size_t)(k0 + row) * N + (n0 + c)]);
        }
        if (k0 + 32 < K) {  // -> global_prefetch_b8
            __builtin_prefetch(&A[(size_t)(m0 + (tid >> 2)) * K + (k0 + 32)], 0, 3);
            __builtin_prefetch(&B[(size_t)(k0 + 32 + (tid >> 3)) * N + n0], 0, 3);
        }
        __syncthreads();

        // ---- fragments ----
        FragB16 af[2], bf[2];
        #pragma unroll
        for (int mi = 0; mi < 2; ++mi) {
            int arow = wm * 32 + mi * 16 + l15;
            #pragma unroll
            for (int j = 0; j < 8; ++j) {   // K pairs: ((j&4)?16:0)+8*hh+2*(j&3)
                int koff = ((j & 4) ? 16 : 0) + 8 * hh + 2 * (j & 3);
                af[mi].w[j] = *reinterpret_cast<const unsigned int*>(&As[arow][koff]);
            }
        }
        #pragma unroll
        for (int ni = 0; ni < 2; ++ni) {
            int bcol = wn * 32 + ni * 16 + l15;
            #pragma unroll
            for (int i = 0; i < 16; ++i) bf[ni].u[i] = Bs[((i & 8) ? 16 : 0) + 8 * hh + (i & 7)][bcol];
        }
        // ---- 4 WMMAs, A/B fragment reuse ----
        #pragma unroll
        for (int mi = 0; mi < 2; ++mi)
            #pragma unroll
            for (int ni = 0; ni < 2; ++ni)
                acc[mi][ni].v = __builtin_amdgcn_wmma_f32_16x16x32_bf16(
                    false, af[mi].v, false, bf[ni].v, (short)0, acc[mi][ni].v,
                    false, false);
        __syncthreads();
    }

    // ---- epilogue: C layout — vgpr r holds (M = r + 8*hh, N = l15) ----
    #pragma unroll
    for (int mi = 0; mi < 2; ++mi) {
        #pragma unroll
        for (int ni = 0; ni < 2; ++ni) {
            #pragma unroll
            for (int r = 0; r < 8; ++r) {
                int m = m0 + wm * 32 + mi * 16 + r + 8 * hh;
                int n = n0 + wn * 32 + ni * 16 + l15;
                float val = acc[mi][ni].f[r];
                if (MODE == 0) {
                    C[(size_t)m * N + n] = val;
                } else {
                    int b = m >> 11, t = m & 2047;
                    int which = n >> 10;
                    int c = n & 1023;
                    int h = c >> 6, d = c & 63;
                    unsigned short* dst = (which == 0) ? Qb : ((which == 1) ? Kb : Vb);
                    dst[((size_t)((b * 16 + h) * 2048 + t)) * 64 + d] = f2bf(val);
                }
            }
        }
    }
}

// --------------------------- flash attention -------------------------------
// Block: 128 threads = 4 waves, covering 64 query rows of one (b,h).
// K/V 32-key tiles staged once in LDS and shared by all 4 waves.
// Online softmax; Dh = 64; scale = 0.125. Waves past their causal bound
// skip compute but keep barrier counts uniform.
__global__ __launch_bounds__(128) void flash_attn_wmma(
    const unsigned short* __restrict__ Qg,  // [B*H, 2048, 64] bf16
    const unsigned short* __restrict__ Kg,
    const unsigned short* __restrict__ Vg,
    unsigned short* __restrict__ Og)        // [B*T, 1024] bf16 (heads merged)
{
    __shared__ unsigned short Ks[32][64];     // K key-block tile (shared)
    __shared__ unsigned short Vs[32][64];     // V key-block tile (shared)
    __shared__ unsigned short Ps[4][16][32];  // per-wave P relayout scratch

    const int tid  = threadIdx.x;
    const int lane = tid & 31;
    const int wv   = tid >> 5;              // 0..3
    const int hh   = lane >> 4;
    const int l15  = lane & 15;
    const int qt64 = blockIdx.x;            // 0..31  (64-row Q super-tiles)
    const int bh   = blockIdx.y;            // 0..31

    const unsigned short* Qh = Qg + (size_t)bh * 2048 * 64;
    const unsigned short* Kh = Kg + (size_t)bh * 2048 * 64;
    const unsigned short* Vh = Vg + (size_t)bh * 2048 * 64;

    const int qrow0 = qt64 * 64 + wv * 16;  // this wave's 16-row tile

    // Q fragments, K-dim halves 0/1 (A layout: row=l15; K=(i>=8?16:0)+8hh+(i&7))
    FragB16 aq0, aq1;
    {
        int qrow = qrow0 + l15;
        const unsigned short* qp = &Qh[(size_t)qrow * 64];
        aq0.q[0] = *reinterpret_cast<const uint4*>(qp + 8 * hh);
        aq0.q[1] = *reinterpret_cast<const uint4*>(qp + 16 + 8 * hh);
        aq1.q[0] = *reinterpret_cast<const uint4*>(qp + 32 + 8 * hh);
        aq1.q[1] = *reinterpret_cast<const uint4*>(qp + 48 + 8 * hh);
    }

    FragF8 o[4];
    #pragma unroll
    for (int d = 0; d < 4; ++d)
        #pragma unroll
        for (int r = 0; r < 8; ++r) o[d].f[r] = 0.f;

    float mrow[8], lrow[8];
    #pragma unroll
    for (int r = 0; r < 8; ++r) { mrow[r] = -1e30f; lrow[r] = 0.f; }

    const int wave_lastrow = qrow0 + 15;
    const int nkb = ((qt64 * 64 + 63) >> 5) + 1;   // uniform per block

    for (int kb = 0; kb < nkb; ++kb) {
        const int kbase = kb * 32;

        // ---- cooperative staging: K,V tiles (each 32x64 bf16 = 256 uint4) ----
        {
            const uint4* ks = reinterpret_cast<const uint4*>(Kh + (size_t)kbase * 64);
            const uint4* vs = reinterpret_cast<const uint4*>(Vh + (size_t)kbase * 64);
            uint4* kd = reinterpret_cast<uint4*>(&Ks[0][0]);
            uint4* vd = reinterpret_cast<uint4*>(&Vs[0][0]);
            kd[tid] = ks[tid];  kd[tid + 128] = ks[tid + 128];
            vd[tid] = vs[tid];  vd[tid + 128] = vs[tid + 128];
        }
        __syncthreads();

        const bool active = (kbase <= wave_lastrow);
        if (active) {
            // ---- S = Q*K^T: two 16x16 f32 tiles (keys +0..15, +16..31) ----
            floatx8 s0 = {0.f,0.f,0.f,0.f,0.f,0.f,0.f,0.f};
            floatx8 s1 = {0.f,0.f,0.f,0.f,0.f,0.f,0.f,0.f};
            #pragma unroll
            for (int sub = 0; sub < 2; ++sub) {
                int keyl = sub * 16 + l15;          // B col = l15
                #pragma unroll
                for (int kk = 0; kk < 2; ++kk) {
                    FragB16 bk;   // K-dim rows kk*32+(i>=8?16:0)+8*hh+(i&7)
                    const uint4* p =
                        reinterpret_cast<const uint4*>(&Ks[keyl][kk * 32 + 8 * hh]);
                    bk.q[0] = p[0]; bk.q[1] = p[2];
                    if (sub == 0)
                        s0 = __builtin_amdgcn_wmma_f32_16x16x32_bf16(
                            false, (kk == 0) ? aq0.v : aq1.v, false, bk.v,
                            (short)0, s0, false, false);
                    else
                        s1 = __builtin_amdgcn_wmma_f32_16x16x32_bf16(
                            false, (kk == 0) ? aq0.v : aq1.v, false, bk.v,
                            (short)0, s1, false, false);
                }
            }

            // ---- causal mask + online softmax ----
            FragF8 u0, u1; u0.v = s0; u1.v = s1;
            #pragma unroll
            for (int r = 0; r < 8; ++r) {
                int row = qrow0 + r + 8 * hh;
                int c0 = kbase + l15;
                int c1 = c0 + 16;
                float f0 = u0.f[r] * 0.125f; if (c0 > row) f0 = -1e30f;
                float f1 = u1.f[r] * 0.125f; if (c1 > row) f1 = -1e30f;

                float tm = fmaxf(f0, f1);
                tm = fmaxf(tm, __shfl_xor(tm, 1, 32));
                tm = fmaxf(tm, __shfl_xor(tm, 2, 32));
                tm = fmaxf(tm, __shfl_xor(tm, 4, 32));
                tm = fmaxf(tm, __shfl_xor(tm, 8, 32));

                float mnew  = fmaxf(mrow[r], tm);
                float alpha = __expf(mrow[r] - mnew);
                float p0 = __expf(f0 - mnew);
                float p1 = __expf(f1 - mnew);

                float rs = p0 + p1;
                rs += __shfl_xor(rs, 1, 32);
                rs += __shfl_xor(rs, 2, 32);
                rs += __shfl_xor(rs, 4, 32);
                rs += __shfl_xor(rs, 8, 32);

                lrow[r] = lrow[r] * alpha + rs;
                mrow[r] = mnew;
                #pragma unroll
                for (int d = 0; d < 4; ++d) o[d].f[r] *= alpha;

                Ps[wv][r + 8 * hh][l15]      = f2bf(p0);
                Ps[wv][r + 8 * hh][16 + l15] = f2bf(p1);
            }
            // same-wave LDS ops are in order: safe to read Ps[wv] now

            // ---- O += P(16x32) * V(32x64): 4 WMMAs ----
            FragB16 ap;
            #pragma unroll
            for (int i = 0; i < 16; ++i) {
                int kk = ((i & 8) ? 16 : 0) + 8 * hh + (i & 7);
                ap.u[i] = Ps[wv][l15][kk];
            }
            #pragma unroll
            for (int db = 0; db < 4; ++db) {
                FragB16 bv;
                #pragma unroll
                for (int i = 0; i < 16; ++i) bv.u[i] = Vs[((i & 8) ? 16 : 0) + 8 * hh + (i & 7)][db * 16 + l15];
                o[db].v = __builtin_amdgcn_wmma_f32_16x16x32_bf16(
                    false, ap.v, false, bv.v, (short)0, o[db].v, false, false);
            }
        }
        __syncthreads();   // protect Ks/Vs before next restaging
    }

    // ---- finalize: normalize, store bf16 into merged-head layout ----
    const int b = bh >> 4, h = bh & 15;
    #pragma unroll
    for (int r = 0; r < 8; ++r) {
        float inv = 1.0f / lrow[r];
        int row_t = qt64 * 64 + wv * 16 + r + 8 * hh;
        #pragma unroll
        for (int db = 0; db < 4; ++db) {
            float val = o[db].f[r] * inv;
            Og[((size_t)(b * 2048 + row_t)) * 1024 + h * 64 + db * 16 + l15] = f2bf(val);
        }
    }
}

// ------------------------------- launcher ----------------------------------
extern "C" void kernel_launch(void* const* d_in, const int* in_sizes, int n_in,
                              void* d_out, int out_size, void* d_ws, size_t ws_size,
                              hipStream_t stream) {
    (void)in_sizes; (void)n_in; (void)out_size; (void)ws_size;
    const float* x     = (const float*)d_in[0];   // [2,2048,1024]
    const float* Wqkv  = (const float*)d_in[1];   // [1024,3072]
    const float* Wout  = (const float*)d_in[2];   // [1024,1024]
    float* out = (float*)d_out;                   // [2,2048,1024]

    const int M = 4096, C = 1024, N3 = 3072;

    // workspace carve-up (all bf16 as ushort), 256B aligned
    char* base = (char*)d_ws;
    size_t off = 0;
    auto alloc = [&](size_t elems) {
        unsigned short* p = (unsigned short*)(base + off);
        off += (elems * sizeof(unsigned short) + 255) & ~(size_t)255;
        return p;
    };
    unsigned short* xb    = alloc((size_t)M * C);      // 8 MB
    unsigned short* wqkvb = alloc((size_t)C * N3);     // 6 MB
    unsigned short* woutb = alloc((size_t)C * C);      // 2 MB
    unsigned short* Qb    = alloc((size_t)M * C);      // 8 MB  [B,H,T,64]
    unsigned short* Kb    = alloc((size_t)M * C);      // 8 MB
    unsigned short* Vb    = alloc((size_t)M * C);      // 8 MB
    unsigned short* attb  = alloc((size_t)M * C);      // 8 MB

    // 1) fp32 -> bf16
    {
        int n = M * C;
        cvt_f32_bf16<<<(n + 255) / 256, 256, 0, stream>>>(x, xb, n);
        n = C * N3;
        cvt_f32_bf16<<<(n + 255) / 256, 256, 0, stream>>>(Wqkv, wqkvb, n);
        n = C * C;
        cvt_f32_bf16<<<(n + 255) / 256, 256, 0, stream>>>(Wout, woutb, n);
    }

    // 2) QKV projection with head-split bf16 scatter (64x128 block tiles)
    gemm_bf16_wmma<1><<<dim3(M / 64, N3 / 128), 256, 0, stream>>>(
        xb, wqkvb, nullptr, Qb, Kb, Vb, M, N3, C);

    // 3) flash attention: 32 Q super-tiles x 32 (b,h), 4 waves/block
    flash_attn_wmma<<<dim3(32, 32), 128, 0, stream>>>(Qb, Kb, Vb, attb);

    // 4) output projection -> fp32 d_out
    gemm_bf16_wmma<0><<<dim3(M / 64, C / 128), 256, 0, stream>>>(
        attb, woutb, out, nullptr, nullptr, nullptr, M, C, C);
}
